// Former_74036646248783
// MI455X (gfx1250) — matrix operand order repaired and checked by reading, not compiled, not run on hardware
//
#include <hip/hip_runtime.h>

typedef __attribute__((ext_vector_type(16))) _Float16 v16h;
typedef __attribute__((ext_vector_type(8)))  _Float16 v8h;
typedef __attribute__((ext_vector_type(8)))  float    v8f;

#define TOKENS 4096   // N*M
#define DMODEL 1024
#define NHEAD  16
#define DHEAD  64
#define SEQ    2048
#define BATCH  2

__device__ __forceinline__ float relu6f(float x) {
    return fminf(fmaxf(x, 0.0f), 6.0f);
}

// Concatenate two 8-half runs into a 16-half WMMA fragment.
__device__ __forceinline__ v16h frag_cat(v8h lo, v8h hi) {
    return __builtin_shufflevector(lo, hi, 0, 1, 2, 3, 4, 5, 6, 7,
                                   8, 9, 10, 11, 12, 13, 14, 15);
}
// A-matrix 16x32 f16 (ISA table): lanes 0-15 K={0..7,16..23}, lanes 16-31 +8;
// per-lane K runs are two contiguous 8-half (16B) chunks at koff and koff+16.
// B-matrix 32x16 f16: same K order as A: lanes 0-15 K={0..7,16..23},
// lanes 16-31 +8; two 8-half (16B) chunks at koff and koff+16.

// DPP16 row_ror rotate within 16-lane rows: pure-VALU cross-lane reductions
// (wave32 halves == DPP rows), replacing ds_bpermute-based __shfl_xor.
template <int CTRL>
__device__ __forceinline__ float dpp_rot(float x) {
    return __int_as_float(__builtin_amdgcn_update_dpp(
        0, __float_as_int(x), CTRL, 0xf, 0xf, true));
}
__device__ __forceinline__ float rowmax16(float x) {
    x = fmaxf(x, dpp_rot<0x128>(x));   // row_ror:8
    x = fmaxf(x, dpp_rot<0x124>(x));   // row_ror:4
    x = fmaxf(x, dpp_rot<0x122>(x));   // row_ror:2
    x = fmaxf(x, dpp_rot<0x121>(x));   // row_ror:1
    return x;
}
__device__ __forceinline__ float rowsum16(float x) {
    x += dpp_rot<0x128>(x);
    x += dpp_rot<0x124>(x);
    x += dpp_rot<0x122>(x);
    x += dpp_rot<0x121>(x);
    return x;
}

// ---------------------------------------------------------------- convert ---
__global__ __launch_bounds__(256) void cvt_f32_to_f16(const float* __restrict__ in,
                                                      _Float16* __restrict__ out, int n) {
    int i = blockIdx.x * 256 + threadIdx.x;
    if (i < n) out[i] = (_Float16)in[i];
}

// Transpose-convert: in f32 [K][N] -> out f16 [N][K] (both sides coalesced).
__global__ __launch_bounds__(256) void tcvt_f32_to_f16(const float* __restrict__ in,
                                                       _Float16* __restrict__ out,
                                                       int K, int N) {
    __shared__ float tile[32][33];
    const int k0 = blockIdx.y * 32, n0 = blockIdx.x * 32;
    const int tx = threadIdx.x, ty = threadIdx.y;       // 32 x 8
    #pragma unroll
    for (int j = 0; j < 4; ++j) {
        int r = ty + j * 8;
        tile[r][tx] = in[(size_t)(k0 + r) * N + n0 + tx];
    }
    __syncthreads();
    #pragma unroll
    for (int j = 0; j < 4; ++j) {
        int r = ty + j * 8;
        out[(size_t)(n0 + r) * K + k0 + tx] = (_Float16)tile[tx][r];
    }
}

// -------------------------------------------------------------------- GEMM ---
// C = oscale * relu6(A[M,K] @ Wt[N,K]^T + bias). A f16 row-major (lda),
// Wt f16 [N][K] row-major (ldb = K). Block tile 128x128, 8 waves of 32x64
// (8 WMMAs per K-step of 32).
// MODE: 0 = f16 row-major, 1 = f32 row-major,
//       2 = f16 head-split [n,h,m,64], 3 = f16 head-split transposed [n,h,64,m]
template <int MODE, bool DO_SCALE>
__global__ __launch_bounds__(256) void gemm_relu6_wmma(
    const _Float16* __restrict__ A, int lda,
    const _Float16* __restrict__ B, int ldb,
    const float* __restrict__ bias,
    _Float16* __restrict__ C16, float* __restrict__ C32,
    int Ndim, int Kdim, float oscale)
{
    __shared__ _Float16 As[128][40];    // 80B rows: 16B-aligned, conflict-free
    __shared__ _Float16 Bst[128][40];   // B tile transposed: [n][k]

    const int tid  = threadIdx.x;
    const int lane = tid & 31;
    const int wid  = tid >> 5;
    const int wr   = wid & 3;             // 0..3 (rows of wave grid)
    const int wc   = wid >> 2;            // 0..1 (cols of wave grid)
    const int bM   = blockIdx.y * 128;
    const int bN   = blockIdx.x * 128;
    const int koff = (lane & 16) ? 8 : 0;   // A-frag K base
    const int boff = (lane & 16) ? 16 : 0;  // B-frag K base (contiguous 16)

    v8f acc[2][4];
    #pragma unroll
    for (int a = 0; a < 2; ++a)
        #pragma unroll
        for (int b = 0; b < 4; ++b) acc[a][b] = (v8f){0,0,0,0,0,0,0,0};

    for (int k0 = 0; k0 < Kdim; k0 += 32) {
        { // stage A 128x32 and Bt 128x32: two v8h each per thread
            int row = tid >> 1, ks = (tid & 1) * 16;
            const _Float16* srcA = A + (size_t)(bM + row) * lda + k0 + ks;
            *(v8h*)&As[row][ks]      = *(const v8h*)(srcA);
            *(v8h*)&As[row][ks + 8]  = *(const v8h*)(srcA + 8);
            const _Float16* srcB = B + (size_t)(bN + row) * ldb + k0 + ks;
            *(v8h*)&Bst[row][ks]     = *(const v8h*)(srcB);
            *(v8h*)&Bst[row][ks + 8] = *(const v8h*)(srcB + 8);
            if (k0 + 32 < Kdim) {        // prefetch next K-tile (global_prefetch)
                __builtin_prefetch(srcA + 32, 0, 1);
                __builtin_prefetch(srcB + 32, 0, 1);
            }
        }
        __syncthreads();

        v16h am[2], bn[4];
        #pragma unroll
        for (int mr = 0; mr < 2; ++mr) {
            int r = wr * 32 + mr * 16 + (lane & 15);
            am[mr] = frag_cat(*(const v8h*)&As[r][koff],
                              *(const v8h*)&As[r][koff + 16]);
        }
        #pragma unroll
        for (int nc = 0; nc < 4; ++nc) {
            int c = wc * 64 + nc * 16 + (lane & 15);
            bn[nc] = frag_cat(*(const v8h*)&Bst[c][koff],
                              *(const v8h*)&Bst[c][koff + 16]);
        }
        #pragma unroll
        for (int mr = 0; mr < 2; ++mr)
            #pragma unroll
            for (int nc = 0; nc < 4; ++nc)
                acc[mr][nc] = __builtin_amdgcn_wmma_f32_16x16x32_f16(
                    false, am[mr], false, bn[nc], (short)0, acc[mr][nc], false, false);
        __syncthreads();
    }

    // epilogue: C frag row = i + 8*(lane>=16), col = lane&15 (branch-free)
    const int half8 = (lane & 16) ? 8 : 0;
    #pragma unroll
    for (int mr = 0; mr < 2; ++mr)
        #pragma unroll
        for (int nc = 0; nc < 4; ++nc) {
            int col = bN + wc * 64 + nc * 16 + (lane & 15);
            float bv = bias[col];
            #pragma unroll
            for (int i = 0; i < 8; ++i) {
                int row = bM + wr * 32 + mr * 16 + i + half8;
                float cv = relu6f(acc[mr][nc][i] + bv);
                if constexpr (DO_SCALE) cv *= oscale;
                if constexpr (MODE == 0) {
                    C16[(size_t)row * Ndim + col] = (_Float16)cv;
                } else if constexpr (MODE == 1) {
                    C32[(size_t)row * Ndim + col] = cv;
                } else {
                    int n = row >> 11, m = row & (SEQ - 1);
                    int h = col & 15,  d = col >> 4;
                    size_t hb = (size_t)(n * NHEAD + h);
                    size_t idx = (MODE == 2) ? (hb * SEQ + m) * DHEAD + d
                                             : (hb * DHEAD + d) * SEQ + m;
                    C16[idx] = (_Float16)cv;
                }
            }
        }
}

// --------------------------------------------------------------- attention ---
// Qs: [n,h,m,64] head-split (pre-scaled by 1/sqrt(dph) in projection);
// Ks: [n,h,m,64]; Vt: [n,h,64,m]. Output v_heads row-major [token][1024],
// channel = h*64+d. Flash online softmax; one wave per 16-query tile; 32-key tiles.
__global__ __launch_bounds__(256) void attn_wmma(
    const _Float16* __restrict__ Qs, const _Float16* __restrict__ Ks,
    const _Float16* __restrict__ Vt, _Float16* __restrict__ Oh)
{
    __shared__ _Float16 P[8][16 * 40];   // per-wave P staging, padded rows

    const int lane = threadIdx.x & 31;
    const int wv   = threadIdx.x >> 5;
    const int gw   = blockIdx.x * 8 + wv;          // 0..4095
    const int qt   = gw & 127;                     // query tile (SEQ/16)
    const int h    = (gw >> 7) & 15;
    const int n    = gw >> 11;
    const int hb   = n * NHEAD + h;                // head-block index
    const int qm   = qt * 16;
    const int half8 = (lane & 16) ? 8 : 0;
    const int koff  = (lane & 16) ? 8 : 0;   // A-frag K base
    const int boff  = (lane & 16) ? 16 : 0;  // B-frag K base (contiguous 16)

    const _Float16* Qbase = Qs + (size_t)hb * SEQ * DHEAD;
    const _Float16* Kbase = Ks + (size_t)hb * SEQ * DHEAD;
    const _Float16* Vbase = Vt + (size_t)hb * DHEAD * SEQ;

    // Q fragments: 16x64 as two 16x32 A-frags (rows contiguous in head-split)
    v16h qa[2];
    {
        const _Float16* qrow = Qbase + (size_t)(qm + (lane & 15)) * DHEAD;
        #pragma unroll
        for (int f = 0; f < 2; ++f)
            qa[f] = frag_cat(*(const v8h*)(qrow + f * 32 + koff),
                             *(const v8h*)(qrow + f * 32 + koff + 16));
    }

    float mrow[8], lrow[8];
    v8f acc[4];
    #pragma unroll
    for (int i = 0; i < 8; ++i) { mrow[i] = -3.0e38f; lrow[i] = 0.0f; }
    #pragma unroll
    for (int ff = 0; ff < 4; ++ff) acc[ff] = (v8f){0,0,0,0,0,0,0,0};

    _Float16* pp = &P[wv][0];

    for (int kt = 0; kt < SEQ / 32; ++kt) {
        const int km = kt * 32;

        // S tile 16x32 = two 16x16 f32 frags (Q pre-scaled => no per-S mul).
        // B = K^T: col = key (lane&15), per-lane K = feats koff..+7, koff+16..+23.
        v8f s[2];
        #pragma unroll
        for (int sub = 0; sub < 2; ++sub) {
            s[sub] = (v8f){0,0,0,0,0,0,0,0};
            const _Float16* krow =
                Kbase + (size_t)(km + sub * 16 + (lane & 15)) * DHEAD;
            #pragma unroll
            for (int f = 0; f < 2; ++f) {
                v16h kf = frag_cat(*(const v8h*)(krow + f * 32 + koff),
                                   *(const v8h*)(krow + f * 32 + koff + 16));
                s[sub] = __builtin_amdgcn_wmma_f32_16x16x32_f16(
                    false, qa[f], false, kf, (short)0, s[sub], false, false);
            }
        }

        // online softmax over 32 keys (DPP row reductions, no LDS traffic)
        #pragma unroll
        for (int i = 0; i < 8; ++i) {
            float mx    = rowmax16(fmaxf(s[0][i], s[1][i]));
            float nm    = fmaxf(mrow[i], mx);
            float alpha = __expf(mrow[i] - nm);
            float p0    = __expf(s[0][i] - nm);
            float p1    = __expf(s[1][i] - nm);
            float sum   = rowsum16(p0 + p1);
            lrow[i] = lrow[i] * alpha + sum;
            mrow[i] = nm;
            #pragma unroll
            for (int ff = 0; ff < 4; ++ff) acc[ff][i] *= alpha;
            int prow = i + half8;
            pp[prow * 40 + (lane & 15)]      = (_Float16)p0;
            pp[prow * 40 + 16 + (lane & 15)] = (_Float16)p1;
        }

        asm volatile("s_wait_dscnt 0" ::: "memory");   // wave-local P write->read

        // P as 16x32 A-frag (two ds_load_b128)
        v16h pa = frag_cat(*(const v8h*)&pp[(lane & 15) * 40 + koff],
                           *(const v8h*)&pp[(lane & 15) * 40 + koff + 16]);

        // acc += P @ V; B = V tile: col = feat d (lane&15), K = keys koff..+7, koff+16..+23
        #pragma unroll
        for (int ff = 0; ff < 4; ++ff) {
            const _Float16* vrow =
                Vbase + (size_t)(ff * 16 + (lane & 15)) * SEQ + km;
            v16h vb = frag_cat(*(const v8h*)(vrow + koff),
                               *(const v8h*)(vrow + koff + 16));
            acc[ff] = __builtin_amdgcn_wmma_f32_16x16x32_f16(
                false, pa, false, vb, (short)0, acc[ff], false, false);
        }
        // (LDS ops are in-order per wave: next-iter P writes cannot pass reads)
    }

    // write v_heads row-major: channel = h*64 + d (head-contiguous)
    float rl[8];
    #pragma unroll
    for (int i = 0; i < 8; ++i) rl[i] = __builtin_amdgcn_rcpf(lrow[i]);
    #pragma unroll
    for (int ff = 0; ff < 4; ++ff) {
        int ch = h * DHEAD + ff * 16 + (lane & 15);
        #pragma unroll
        for (int i = 0; i < 8; ++i) {
            int row = n * SEQ + qm + i + half8;
            Oh[(size_t)row * DMODEL + ch] = (_Float16)(acc[ff][i] * rl[i]);
        }
    }
}

// --------------------------------------------------------------- layernorm ---
template <bool OUT16>
__global__ __launch_bounds__(256) void layernorm_k(const float* __restrict__ X,
                                                   const float* __restrict__ w,
                                                   const float* __restrict__ b,
                                                   void* __restrict__ outp)
{
    __shared__ float red[256];
    const int row = blockIdx.x;
    const int tid = threadIdx.x;
    const float* x = X + (size_t)row * DMODEL;

    float v[4];
    float s = 0.0f;
    #pragma unroll
    for (int j = 0; j < 4; ++j) { v[j] = x[tid + j * 256]; s += v[j]; }
    red[tid] = s; __syncthreads();
    for (int st = 128; st > 0; st >>= 1) {
        if (tid < st) red[tid] += red[tid + st];
        __syncthreads();
    }
    float mu = red[0] * (1.0f / DMODEL);
    __syncthreads();

    float vs = 0.0f;
    #pragma unroll
    for (int j = 0; j < 4; ++j) { float d = v[j] - mu; vs += d * d; }
    red[tid] = vs; __syncthreads();
    for (int st = 128; st > 0; st >>= 1) {
        if (tid < st) red[tid] += red[tid + st];
        __syncthreads();
    }
    float rstd = rsqrtf(red[0] * (1.0f / DMODEL) + 1e-5f);

    #pragma unroll
    for (int j = 0; j < 4; ++j) {
        int c = tid + j * 256;
        float y = (v[j] - mu) * rstd * w[c] + b[c];
        size_t idx = (size_t)row * DMODEL + c;
        if constexpr (OUT16) ((_Float16*)outp)[idx] = (_Float16)y;
        else                 ((float*)outp)[idx]    = y;
    }
}

// ------------------------------------------------------------------ launch ---
extern "C" void kernel_launch(void* const* d_in, const int* in_sizes, int n_in,
                              void* d_out, int out_size, void* d_ws, size_t ws_size,
                              hipStream_t stream) {
    const float* x      = (const float*)d_in[0];
    const float* qvk_w  = (const float*)d_in[1];
    const float* qvk_b  = (const float*)d_in[2];
    const float* qh_w   = (const float*)d_in[3];
    const float* qh_b   = (const float*)d_in[4];
    const float* kh_w   = (const float*)d_in[5];
    const float* kh_b   = (const float*)d_in[6];
    const float* vh_w   = (const float*)d_in[7];
    const float* vh_b   = (const float*)d_in[8];
    const float* o_w    = (const float*)d_in[9];
    const float* o_b    = (const float*)d_in[10];
    const float* ln1_w  = (const float*)d_in[11];
    const float* ln1_b  = (const float*)d_in[12];
    const float* m1_w   = (const float*)d_in[13];
    const float* m1_b   = (const float*)d_in[14];
    const float* m2_w   = (const float*)d_in[15];
    const float* m2_b   = (const float*)d_in[16];
    const float* ln2_w  = (const float*)d_in[17];
    const float* ln2_b  = (const float*)d_in[18];
    float* out = (float*)d_out;

    const size_t MB = 1ull << 20;
    char* base = (char*)d_ws;
    _Float16* xh   = (_Float16*)(base + 0);          //  8 MB [4096,1024]
    _Float16* wbuf = (_Float16*)(base + 8 * MB);     //  6 MB Wt staging (reused)
    _Float16* qvkh = (_Float16*)(base + 14 * MB);    // 24 MB [4096,3072]
    _Float16* Qs   = (_Float16*)(base + 38 * MB);    //  8 MB head-split
    _Float16* Ks   = (_Float16*)(base + 46 * MB);    //  8 MB head-split
    _Float16* Vt   = (_Float16*)(base + 54 * MB);    //  8 MB head-split transposed
    // reuse (stream-ordered, lifetimes disjoint):
    _Float16* vheads = xh;                           // after stage 1 consumed
    float*    attf32 = (float*)qvkh;                 // 16 MB, after QKV done
    _Float16* ln1h   = Qs;                           // after attention
    _Float16* h1h    = Ks;                           // 16 MB spans Ks+Vt
    float*    m2f32  = (float*)qvkh;                 // after LN1 consumed

    dim3 blk(256);
    auto cvt = [&](const float* src, _Float16* dst, int n) {
        cvt_f32_to_f16<<<dim3((n + 255) / 256), blk, 0, stream>>>(src, dst, n);
    };
    auto tcvt = [&](const float* src, _Float16* dst, int K, int N) {
        tcvt_f32_to_f16<<<dim3(N / 32, K / 32), dim3(32, 8), 0, stream>>>(src, dst, K, N);
    };

    // 1) activations + qvk projection (f16 row-major out)
    cvt(x, xh, TOKENS * DMODEL);
    tcvt(qvk_w, wbuf, DMODEL, 3 * DMODEL);
    gemm_relu6_wmma<0, false><<<dim3(3 * DMODEL / 128, TOKENS / 128), blk, 0, stream>>>(
        xh, DMODEL, wbuf, DMODEL, qvk_b, qvkh, nullptr, 3 * DMODEL, DMODEL, 1.0f);

    // 2) per-head projections from qvkh slices (lda = 3072);
    //    Q pre-scaled by 1/sqrt(dph) so attention needs no per-score mul
    tcvt(qh_w, wbuf, DMODEL, DMODEL);
    gemm_relu6_wmma<2, true><<<dim3(DMODEL / 128, TOKENS / 128), blk, 0, stream>>>(
        qvkh + 0 * DMODEL, 3 * DMODEL, wbuf, DMODEL, qh_b, Qs, nullptr, DMODEL, DMODEL, 0.125f);
    tcvt(kh_w, wbuf, DMODEL, DMODEL);
    gemm_relu6_wmma<2, false><<<dim3(DMODEL / 128, TOKENS / 128), blk, 0, stream>>>(
        qvkh + 1 * DMODEL, 3 * DMODEL, wbuf, DMODEL, kh_b, Ks, nullptr, DMODEL, DMODEL, 1.0f);
    tcvt(vh_w, wbuf, DMODEL, DMODEL);
    gemm_relu6_wmma<3, false><<<dim3(DMODEL / 128, TOKENS / 128), blk, 0, stream>>>(
        qvkh + 2 * DMODEL, 3 * DMODEL, wbuf, DMODEL, vh_b, Vt, nullptr, DMODEL, DMODEL, 1.0f);

    // 3) attention (4096 wave-tiles, 8 waves/block)
    attn_wmma<<<dim3(BATCH * NHEAD * (SEQ / 16) / 8), blk, 0, stream>>>(Qs, Ks, Vt, vheads);

    // 4) output projection -> f32 for LN1
    tcvt(o_w, wbuf, DMODEL, DMODEL);
    gemm_relu6_wmma<1, false><<<dim3(DMODEL / 128, TOKENS / 128), blk, 0, stream>>>(
        vheads, DMODEL, wbuf, DMODEL, o_b, nullptr, attf32, DMODEL, DMODEL, 1.0f);
    layernorm_k<true><<<dim3(TOKENS), blk, 0, stream>>>(attf32, ln1_w, ln1_b, ln1h);

    // 5) MLP
    tcvt(m1_w, wbuf, DMODEL, 2 * DMODEL);
    gemm_relu6_wmma<0, false><<<dim3(2 * DMODEL / 128, TOKENS / 128), blk, 0, stream>>>(
        ln1h, DMODEL, wbuf, DMODEL, m1_b, h1h, nullptr, 2 * DMODEL, DMODEL, 1.0f);
    tcvt(m2_w, wbuf, 2 * DMODEL, DMODEL);
    gemm_relu6_wmma<1, false><<<dim3(DMODEL / 128, TOKENS / 128), blk, 0, stream>>>(
        h1h, 2 * DMODEL, wbuf, 2 * DMODEL, m2_b, nullptr, m2f32, DMODEL, 2 * DMODEL, 1.0f);

    // 6) final layernorm -> fp32 output
    layernorm_k<false><<<dim3(TOKENS), blk, 0, stream>>>(m2f32, ln2_w, ln2_b, out);
}
